// LSTMGC_27951647162603
// MI455X (gfx1250) — matrix operand order repaired and checked by reading, not compiled, not run on hardware
//
#include <hip/hip_runtime.h>

// ---------------------------------------------------------------------------
// Problem constants (from reference setup_inputs)
// ---------------------------------------------------------------------------
#define B_      32
#define T_      24
#define N_      325
#define F_      32      // input features
#define OUT_    64      // gcn output per branch
#define FEAT_   128     // 2*OUT_
#define UNITS_  256
#define G4_     1024    // 4*UNITS_
#define HOR_    12
#define MROWS_  (B_ * N_)          // 10400 LSTM rows
#define XROWS_  (MROWS_ * T_)      // 249600 gcn output rows
#define NBT_    (N_ * B_ * T_)     // aggregate threads

typedef unsigned short u16;
typedef __attribute__((ext_vector_type(16))) __bf16 v16bf;
typedef __attribute__((ext_vector_type(8)))  float  v8f;
typedef __attribute__((ext_vector_type(4)))  unsigned int u32x4;
typedef __attribute__((ext_vector_type(8)))  int i32x8;
typedef __attribute__((ext_vector_type(4)))  int i32x4;

union FragAB {
    v16bf v;
    uint4 q[2];
    u16   s[16];
    __bf16 b[16];
};

static __device__ __forceinline__ u16 f2bf(float f) {
    __bf16 h = (__bf16)f;                       // hardware RNE convert
    return __builtin_bit_cast(u16, h);
}
static __device__ __forceinline__ float bf2f(u16 s) {
    union { unsigned u; float f; } x; x.u = ((unsigned)s) << 16;
    return x.f;
}
static __device__ __forceinline__ float sigm(float x) {
    return 1.0f / (1.0f + __expf(-x));
}
static __device__ __forceinline__ v8f wmma_bf16(v16bf a, v16bf b, v8f c) {
    // D = A(16x32 bf16) * B(32x16 bf16) + C(16x16 f32)
    return __builtin_amdgcn_wmma_f32_16x16x32_bf16(
        /*neg_a=*/false, a, /*neg_b=*/false, b,
        /*c_mod=*/(short)0, c, /*reuse_a=*/false, /*reuse_b=*/false);
}

// ---------------------------------------------------------------------------
// Kernel 1: deterministic CSR build (single thread; E ~ 2600, trivial cost)
// edges[0..E)   = destination (segment) node
// edges[E..2E)  = source (neighbor) node
// ---------------------------------------------------------------------------
__global__ void build_csr_kernel(const int* __restrict__ edges, int E,
                                 int* __restrict__ csr_off,
                                 int* __restrict__ csr_fill,
                                 int* __restrict__ csr_src) {
    if (threadIdx.x != 0 || blockIdx.x != 0) return;
    for (int n = 0; n <= N_; ++n) csr_off[n] = 0;
    for (int e = 0; e < E; ++e) csr_off[edges[e] + 1] += 1;
    for (int n = 1; n <= N_; ++n) csr_off[n] += csr_off[n - 1];
    for (int n = 0; n < N_; ++n) csr_fill[n] = csr_off[n];
    for (int e = 0; e < E; ++e) {
        int d = edges[e];
        csr_src[csr_fill[d]++] = edges[E + e];
    }
}

// ---------------------------------------------------------------------------
// Kernel 2: f32 [R][C] -> bf16 transposed [C][R]
// ---------------------------------------------------------------------------
__global__ void transpose_bf16_kernel(const float* __restrict__ src,
                                      u16* __restrict__ dst, int R, int C) {
    int tid = blockIdx.x * blockDim.x + threadIdx.x;
    if (tid >= R * C) return;
    int r = tid / C, c = tid % C;
    dst[c * R + r] = f2bf(src[r * C + c]);
}

// ---------------------------------------------------------------------------
// Kernel 3: neighbor-mean aggregation (deterministic CSR order)
// mean layout: [node][b][t][f]  (f32)
// ---------------------------------------------------------------------------
__global__ void aggregate_kernel(const float* __restrict__ feats,
                                 const int* __restrict__ csr_off,
                                 const int* __restrict__ csr_src,
                                 float* __restrict__ mean) {
    int tid = blockIdx.x * blockDim.x + threadIdx.x;
    if (tid >= NBT_) return;
    int node = tid / (B_ * T_);
    int bt   = tid % (B_ * T_);
    int bb = bt / T_, tt = bt % T_;
    int beg = csr_off[node], end = csr_off[node + 1];
    float acc[F_];
#pragma unroll
    for (int f = 0; f < F_; ++f) acc[f] = 0.0f;
    for (int e = beg; e < end; ++e) {
        int src = csr_src[e];
        const float4* sp = reinterpret_cast<const float4*>(
            feats + (((size_t)(bb * T_ + tt)) * N_ + src) * F_);
#pragma unroll
        for (int q = 0; q < F_ / 4; ++q) {
            float4 v = sp[q];
            acc[q * 4 + 0] += v.x; acc[q * 4 + 1] += v.y;
            acc[q * 4 + 2] += v.z; acc[q * 4 + 3] += v.w;
        }
    }
    int cnt = end - beg;
    float inv = 1.0f / (float)(cnt > 1 ? cnt : 1);
    float4* mp = reinterpret_cast<float4*>(
        mean + (((size_t)(node * B_ + bb)) * T_ + tt) * F_);
#pragma unroll
    for (int q = 0; q < F_ / 4; ++q) {
        float4 v;
        v.x = acc[q * 4 + 0] * inv; v.y = acc[q * 4 + 1] * inv;
        v.z = acc[q * 4 + 2] * inv; v.w = acc[q * 4 + 3] * inv;
        mp[q] = v;
    }
}

// ---------------------------------------------------------------------------
// Kernel 4: GCN projection via WMMA.
//   x[r, 0:64]    = relu(feats_row  @ W_g)
//   x[r, 64:128]  = relu(mean_row   @ W_g)
// Row ordering r = ((b*N + n)*T + t) -> directly the LSTM input layout.
// K = 32 == one bf16 WMMA k-step. 8 waves/block, one 16-row tile per wave.
// WgT: bf16 [OUT_][F_] (transposed so B-frags are contiguous).
// ---------------------------------------------------------------------------
__global__ __launch_bounds__(256)
void gcn_kernel(const float* __restrict__ feats,
                const float* __restrict__ mean,
                const u16* __restrict__ WgT,
                u16* __restrict__ xbf) {
    int wave = threadIdx.x >> 5, lane = threadIdx.x & 31;
    int nn = lane & 15, hh = lane >> 4;
    int rtile = blockIdx.x * 8 + wave;
    int r0 = rtile * 16;
    int r = r0 + nn;                       // A-frag row of this lane
    int bb   = r / (N_ * T_);
    int rem  = r % (N_ * T_);
    int node = rem / T_;
    int tt   = rem % T_;

    const float* frow = feats + (((size_t)(bb * T_ + tt)) * N_ + node) * F_;
    const float* mrow = mean  + (((size_t)(node * B_ + bb)) * T_ + tt) * F_;

    // B fragments: 4 column tiles of W_g, shared by both branches.
    FragAB bfr[4];
#pragma unroll
    for (int ct = 0; ct < 4; ++ct) {
        const u16* wp = WgT + (size_t)(ct * 16 + nn) * F_ + 8 * hh;
        bfr[ct].q[0] = *reinterpret_cast<const uint4*>(wp);
        bfr[ct].q[1] = *reinterpret_cast<const uint4*>(wp + 16);
    }

#pragma unroll
    for (int part = 0; part < 2; ++part) {
        const float* src = part ? mrow : frow;
        float4 p0 = *reinterpret_cast<const float4*>(src + 8 * hh);
        float4 p1 = *reinterpret_cast<const float4*>(src + 8 * hh + 4);
        float4 p2 = *reinterpret_cast<const float4*>(src + 16 + 8 * hh);
        float4 p3 = *reinterpret_cast<const float4*>(src + 20 + 8 * hh);
        FragAB a;
        a.b[0]  = (__bf16)p0.x; a.b[1]  = (__bf16)p0.y; a.b[2]  = (__bf16)p0.z; a.b[3]  = (__bf16)p0.w;
        a.b[4]  = (__bf16)p1.x; a.b[5]  = (__bf16)p1.y; a.b[6]  = (__bf16)p1.z; a.b[7]  = (__bf16)p1.w;
        a.b[8]  = (__bf16)p2.x; a.b[9]  = (__bf16)p2.y; a.b[10] = (__bf16)p2.z; a.b[11] = (__bf16)p2.w;
        a.b[12] = (__bf16)p3.x; a.b[13] = (__bf16)p3.y; a.b[14] = (__bf16)p3.z; a.b[15] = (__bf16)p3.w;

#pragma unroll
        for (int ct = 0; ct < 4; ++ct) {
            v8f c;
#pragma unroll
            for (int v = 0; v < 8; ++v) c[v] = 0.0f;
            c = wmma_bf16(a.v, bfr[ct].v, c);
            int col = part * OUT_ + ct * 16 + nn;
#pragma unroll
            for (int v = 0; v < 8; ++v) {
                float x = c[v] > 0.0f ? c[v] : 0.0f;
                xbf[(size_t)(r0 + 8 * hh + v) * FEAT_ + col] = f2bf(x);
            }
        }
    }
}

// ---------------------------------------------------------------------------
// Kernel 5: fused LSTM (24 steps) + dense head. One block per 16-row tile.
// 16 waves; wave w owns unit-column tile w (16 units) and accumulates the
// four gate tiles (i,f,g,o) for it. The x-tile is DMA'd into LDS by the
// Tensor Data Mover (3D tile: x=k(128), y=m(16, stride 3072), z=t(24,
// stride 128)) with hardware padding of 4 DWORDs every 64 DWORDs, which
// reproduces the bank-spread [t][m][XSTRIDE=136] layout exactly.
// h ping-pongs through LDS (bf16), c stays in fp32 registers.
// WlT: bf16 [G4_][FEAT_]; UT: bf16 [G4_][UNITS_]  (both transposed)
// ---------------------------------------------------------------------------
#define XSTRIDE 136   // 128 + 8 pad (16B aligned rows, spreads banks)
#define HSTRIDE 264   // 256 + 8 pad

__global__ __launch_bounds__(512)
void lstm_dense_kernel(const u16* __restrict__ xbf,
                       const u16* __restrict__ WlT,
                       const u16* __restrict__ UT,
                       const float* __restrict__ b_lstm,
                       const float* __restrict__ W_d,
                       const float* __restrict__ b_d,
                       float* __restrict__ out) {
    __shared__ __align__(16) u16 xlds[T_ * 16 * XSTRIDE];  // ~102 KB
    __shared__ __align__(16) u16 hlds[16 * HSTRIDE];       // ~8 KB

    int tid  = threadIdx.x;
    int wave = tid >> 5, lane = tid & 31;
    int nn = lane & 15, hh = lane >> 4;
    int r0 = blockIdx.x * 16;

    // ---- TDM: DMA the 16x24x128 bf16 x-tile into LDS (wave 0 issues).
    if (wave == 0) {
        unsigned lds_addr = (unsigned)(uintptr_t)(&xlds[0]);
        unsigned long long ga =
            (unsigned long long)(uintptr_t)(xbf + (size_t)r0 * (T_ * FEAT_));
        u32x4 g0;
        g0[0] = 1u;                                   // count = 1 descriptor
        g0[1] = lds_addr;                             // lds_addr [63:32]
        g0[2] = (unsigned)(ga & 0xFFFFFFFFull);       // global_addr low
        g0[3] = (unsigned)((ga >> 32) & 0x01FFFFFFull) | (2u << 30); // hi + type=2
        i32x8 g1;
        g1[0] = (int)((1u << 16)        // data_size = 1 -> 2 bytes
                    | (1u << 20)        // pad_enable
                    | (5u << 22)        // pad_interval: every 64 DWORDs (256B)
                    | (3u << 25));      // pad_amount: 4 DWORDs (16B)
        g1[1] = (int)(128u << 16);      // tensor_dim0 = 128 (low 16)
        g1[2] = (int)(16u << 16);       // tensor_dim0 hi=0 | tensor_dim1 = 16
        g1[3] = (int)(128u << 16);      // tensor_dim1 hi=0 | tile_dim0 = 128
        g1[4] = (int)(16u | (24u << 16)); // tile_dim1 = 16, tile_dim2 = 24
        g1[5] = (int)3072;              // tensor_dim0_stride = 3072 (m stride)
        g1[6] = (int)(128u << 16);      // d0_stride hi=0 | dim1_stride = 128 (t)
        g1[7] = 0;                      // dim1_stride hi
        i32x4 g2;
        g2[0] = 24;                     // tensor_dim2 = 24
        g2[1] = 0; g2[2] = 0; g2[3] = 0;
        i32x4 g3;
        g3[0] = 0; g3[1] = 0; g3[2] = 0; g3[3] = 0;
        i32x8 gz;                       // extra SGPR group (6-arg form): zero
        gz[0] = 0; gz[1] = 0; gz[2] = 0; gz[3] = 0;
        gz[4] = 0; gz[5] = 0; gz[6] = 0; gz[7] = 0;
        __builtin_amdgcn_tensor_load_to_lds(g0, g1, g2, g3, gz, 0);
    }
    for (int i = tid; i < 16 * HSTRIDE; i += 512) hlds[i] = 0;  // h_{-1} = 0
    if (wave == 0) __builtin_amdgcn_s_wait_tensorcnt(0);
    __syncthreads();

    int ucol = wave * 16 + nn;                     // unit column of this lane
    float bias[4];
#pragma unroll
    for (int g = 0; g < 4; ++g) bias[g] = b_lstm[g * UNITS_ + ucol];

    const u16* wl[4];
    const u16* uu[4];
#pragma unroll
    for (int g = 0; g < 4; ++g) {
        wl[g] = WlT + (size_t)(g * UNITS_ + ucol) * FEAT_;
        uu[g] = UT  + (size_t)(g * UNITS_ + ucol) * UNITS_;
        __builtin_prefetch(wl[g], 0, 0);
        __builtin_prefetch(uu[g], 0, 0);
    }

    float cst[8];
#pragma unroll
    for (int v = 0; v < 8; ++v) cst[v] = 0.0f;

    for (int t = 0; t < T_; ++t) {
        v8f z[4];
#pragma unroll
        for (int g = 0; g < 4; ++g)
#pragma unroll
            for (int v = 0; v < 8; ++v) z[g][v] = bias[g];

        // ---- input projection: z += x_t @ W_lstm   (K = 128, 4 k-steps)
        const u16* xrow = &xlds[(t * 16 + nn) * XSTRIDE];
#pragma unroll
        for (int kk = 0; kk < 4; ++kk) {
            int kb = 32 * kk;
            FragAB a;
            a.q[0] = *reinterpret_cast<const uint4*>(&xrow[kb + 8 * hh]);
            a.q[1] = *reinterpret_cast<const uint4*>(&xrow[kb + 16 + 8 * hh]);
#pragma unroll
            for (int g = 0; g < 4; ++g) {
                FragAB b;
                const u16* wp = wl[g] + kb + 8 * hh;
                b.q[0] = *reinterpret_cast<const uint4*>(wp);
                b.q[1] = *reinterpret_cast<const uint4*>(wp + 16);
                z[g] = wmma_bf16(a.v, b.v, z[g]);
            }
        }

        // ---- recurrence: z += h_{t-1} @ U   (K = 256, 8 k-steps)
        const u16* hrow = &hlds[nn * HSTRIDE];
#pragma unroll
        for (int kk = 0; kk < 8; ++kk) {
            int kb = 32 * kk;
            FragAB a;
            a.q[0] = *reinterpret_cast<const uint4*>(&hrow[kb + 8 * hh]);
            a.q[1] = *reinterpret_cast<const uint4*>(&hrow[kb + 16 + 8 * hh]);
#pragma unroll
            for (int g = 0; g < 4; ++g) {
                FragAB b;
                const u16* up = uu[g] + kb + 8 * hh;
                b.q[0] = *reinterpret_cast<const uint4*>(up);
                b.q[1] = *reinterpret_cast<const uint4*>(up + 16);
                z[g] = wmma_bf16(a.v, b.v, z[g]);
            }
        }
        __syncthreads();   // all hlds reads complete before overwrite

        // ---- gates (fp32), cell update, write h_t (bf16) to LDS
#pragma unroll
        for (int v = 0; v < 8; ++v) {
            float iv = sigm(z[0][v]);
            float fv = sigm(z[1][v]);
            float gv = z[2][v] > 0.0f ? z[2][v] : 0.0f;
            float ov = sigm(z[3][v]);
            cst[v] = fv * cst[v] + iv * gv;
            float rc = cst[v] > 0.0f ? cst[v] : 0.0f;
            hlds[(8 * hh + v) * HSTRIDE + ucol] = f2bf(ov * rc);
        }
        __syncthreads();
    }

    // ---- dense head: out[b, j, n] = h @ W_d + b_d
    if (tid < 16 * HOR_) {
        int rl = tid / HOR_, j = tid % HOR_;
        float acc = b_d[j];
        const u16* hp = &hlds[rl * HSTRIDE];
#pragma unroll 8
        for (int u = 0; u < UNITS_; ++u)
            acc += bf2f(hp[u]) * W_d[u * HOR_ + j];
        int r = r0 + rl;
        int bb = r / N_, node = r % N_;
        out[((size_t)bb * HOR_ + j) * N_ + node] = acc;
    }
}

// ---------------------------------------------------------------------------
// Host launcher
// ---------------------------------------------------------------------------
static inline size_t al256(size_t x) { return (x + 255) & ~(size_t)255; }

extern "C" void kernel_launch(void* const* d_in, const int* in_sizes, int n_in,
                              void* d_out, int out_size, void* d_ws, size_t ws_size,
                              hipStream_t stream) {
    const float* feats  = (const float*)d_in[0];   // [B,T,N,F]
    const int*   edges  = (const int*)  d_in[1];   // [2,E]
    const float* W_g    = (const float*)d_in[2];   // [F,OUT]
    const float* W_lstm = (const float*)d_in[3];   // [FEAT,4U]
    const float* U_lstm = (const float*)d_in[4];   // [U,4U]
    const float* b_lstm = (const float*)d_in[5];   // [4U]
    const float* W_d    = (const float*)d_in[6];   // [U,HOR]
    const float* b_d    = (const float*)d_in[7];   // [HOR]
    float*       out    = (float*)d_out;           // [B,HOR,N]

    const int E = in_sizes[1] / 2;

    // Workspace carve-up
    char* ws = (char*)d_ws;
    size_t o = 0;
    size_t o_mean   = o; o = al256(o + (size_t)NBT_ * F_ * sizeof(float));
    size_t o_csroff = o; o = al256(o + (size_t)(N_ + 1) * sizeof(int));
    size_t o_csrfil = o; o = al256(o + (size_t)N_ * sizeof(int));
    size_t o_csrsrc = o; o = al256(o + (size_t)E * sizeof(int));
    size_t o_wgt    = o; o = al256(o + (size_t)OUT_ * F_ * sizeof(u16));
    size_t o_wlt    = o; o = al256(o + (size_t)G4_ * FEAT_ * sizeof(u16));
    size_t o_ut     = o; o = al256(o + (size_t)G4_ * UNITS_ * sizeof(u16));
    size_t o_xbf    = o; o = al256(o + (size_t)XROWS_ * FEAT_ * sizeof(u16));
    if (o > ws_size) return;  // workspace too small: do nothing (deterministic)

    float* mean    = (float*)(ws + o_mean);
    int*   csr_off = (int*)  (ws + o_csroff);
    int*   csr_fil = (int*)  (ws + o_csrfil);
    int*   csr_src = (int*)  (ws + o_csrsrc);
    u16*   WgT     = (u16*)  (ws + o_wgt);
    u16*   WlT     = (u16*)  (ws + o_wlt);
    u16*   UT      = (u16*)  (ws + o_ut);
    u16*   xbf     = (u16*)  (ws + o_xbf);

    // 1. deterministic CSR
    build_csr_kernel<<<1, 64, 0, stream>>>(edges, E, csr_off, csr_fil, csr_src);

    // 2. weight transpose + bf16 conversion
    transpose_bf16_kernel<<<(F_ * OUT_ + 255) / 256, 256, 0, stream>>>(W_g, WgT, F_, OUT_);
    transpose_bf16_kernel<<<(FEAT_ * G4_ + 255) / 256, 256, 0, stream>>>(W_lstm, WlT, FEAT_, G4_);
    transpose_bf16_kernel<<<(UNITS_ * G4_ + 255) / 256, 256, 0, stream>>>(U_lstm, UT, UNITS_, G4_);

    // 3. neighbor mean
    aggregate_kernel<<<(NBT_ + 255) / 256, 256, 0, stream>>>(feats, csr_off, csr_src, mean);

    // 4. GCN projection (WMMA): 15600 row-tiles, 8 waves/block
    gcn_kernel<<<XROWS_ / (16 * 8), 256, 0, stream>>>(feats, mean, WgT, xbf);

    // 5. fused LSTM + dense head: one block per 16-row tile
    lstm_dense_kernel<<<MROWS_ / 16, 512, 0, stream>>>(xbf, WlT, UT, b_lstm, W_d, b_d, out);
}
